// RGATLayer_81552839016948
// MI455X (gfx1250) — compile-verified, matrix operand order checked by reading, not run on hardware
//
#include <hip/hip_runtime.h>
#include <hip/hip_bf16.h>

#define N_NODES 65536
#define IN_FEAT 512
#define OUT_FEAT 256
#define N_REL 3
#define N_EDGES 500000
#define NEG_SLOPE 0.2f
#define EPS 1e-9f

typedef unsigned short u16;
typedef __attribute__((ext_vector_type(16))) __bf16 v16bf;
typedef __attribute__((ext_vector_type(8)))  __bf16 v8bf;
typedef __attribute__((ext_vector_type(8)))  float  v8f;
typedef int v4i __attribute__((vector_size(16)));   // matches builtin param type

#define AS1 __attribute__((address_space(1)))
#define AS3 __attribute__((address_space(3)))

#define BM 128
#define BN 64
#define BK 32
#define LDA 40          // padded LDS row stride in bf16 elems (80B, 16B-aligned)

#if __has_builtin(__builtin_amdgcn_global_load_async_to_lds_b128)
#define HAVE_ASYNC_LDS 1
#else
#define HAVE_ASYNC_LDS 0
#endif

// 16-byte global->LDS copy: CDNA5 async-to-LDS path when available.
__device__ __forceinline__ void copy16_g2l(const u16* g, u16* l) {
#if HAVE_ASYNC_LDS
    AS1 u16* g1 = (AS1 u16*)(u16*)g;     // addrspacecast (generic -> global)
    AS3 u16* l3 = (AS3 u16*)l;           // addrspacecast (generic -> LDS)
    __builtin_amdgcn_global_load_async_to_lds_b128(
        (AS1 v4i*)g1, (AS3 v4i*)l3, 0, 0);
#else
    *(uint4*)l = *(const uint4*)g;
#endif
}

__device__ __forceinline__ void wait_async_lds() {
#if HAVE_ASYNC_LDS
#if __has_builtin(__builtin_amdgcn_s_wait_asynccnt)
    __builtin_amdgcn_s_wait_asynccnt(0);
#else
    asm volatile("s_wait_asynccnt 0" ::: "memory");
#endif
#endif
}

#define CAT16(lo, hi8) __builtin_shufflevector(lo, hi8, 0,1,2,3,4,5,6,7,8,9,10,11,12,13,14,15)

// ---------------------------------------------------------------------------
// One-shot precision conversion kernels.
// ---------------------------------------------------------------------------
// h [N,512] f32 -> bf16 (pairwise packed stores)
__global__ __launch_bounds__(256) void convert_h_kernel(
    const float* __restrict__ H, u16* __restrict__ Hbf)
{
    int i = blockIdx.x * blockDim.x + threadIdx.x;     // pair index
    float2 v = ((const float2*)H)[i];
    unsigned int lo = __builtin_bit_cast(u16, (__bf16)v.x);
    unsigned int hi = __builtin_bit_cast(u16, (__bf16)v.y);
    ((unsigned int*)Hbf)[i] = lo | (hi << 16);
}

// W [rel][K][N] f32 -> WbfT [rel][N][K] bf16 (transposed for row-contig B tiles)
__global__ __launch_bounds__(256) void convert_wT_kernel(
    const float* __restrict__ W, u16* __restrict__ WT)
{
    int i = blockIdx.x * blockDim.x + threadIdx.x;     // over N_REL*IN_FEAT*OUT_FEAT
    int n   = i & (OUT_FEAT - 1);
    int rk  = i >> 8;                                  // rel*512 + k
    int k   = rk & (IN_FEAT - 1);
    int rel = rk >> 9;
    WT[((size_t)(rel * OUT_FEAT + n)) * IN_FEAT + k] =
        __builtin_bit_cast(u16, (__bf16)W[i]);
}

// ---------------------------------------------------------------------------
// GEMM: Z = h @ W[rel]  (M=65536, K=512, N=256), bf16 WMMA, f32 accumulate.
// Block tile 128x64, BK=32, double-buffered bf16 LDS, async global->LDS.
// 8 waves: wave grid 4(M)x2(N), each wave = 32x32 tile = 4 WMMA / K-step,
// fragments = 2x ds_load_b128 each (ISA 7.12.2 half-wave K layout).
// ---------------------------------------------------------------------------
__global__ __launch_bounds__(256) void gemm_bf16_wmma(
    const u16* __restrict__ Hbf, const u16* __restrict__ WbfT,
    float* __restrict__ Z, int rel)
{
    __shared__ __align__(16) u16 sA[2][BM][LDA];       // 20,480 B
    __shared__ __align__(16) u16 sB[2][BN][LDA];       // 10,240 B

    const int tid  = threadIdx.x;
    const int lane = tid & 31;
    const int wave = tid >> 5;
    const int m0   = blockIdx.y * BM;
    const int n0   = blockIdx.x * BN;
    const int wm   = (wave & 3) << 5;
    const int wn   = (wave >> 2) << 5;

    const u16* __restrict__ Wr = WbfT + (size_t)rel * OUT_FEAT * IN_FEAT;

    // staging assignment: 16B chunks; A tile = 512 chunks (2/thread), B = 256
    const int arow0 = tid >> 2;               // rows 0..63
    const int arow1 = arow0 + 64;             // rows 64..127
    const int aseg  = (tid & 3) << 3;         // K sub-offset {0,8,16,24}
    const int brow  = tid >> 2;               // B rows (N) 0..63

    const u16* gA0 = Hbf + (size_t)(m0 + arow0) * IN_FEAT + aseg;
    const u16* gA1 = Hbf + (size_t)(m0 + arow1) * IN_FEAT + aseg;
    const u16* gB  = Wr  + (size_t)(n0 + brow) * IN_FEAT + aseg;

    v8f c00 = {}, c01 = {}, c10 = {}, c11 = {};
    const int lr = lane & 15;      // row (A) / col (B) within fragment
    const int hi = lane >> 4;      // half-wave
    const int kb = hi << 3;        // K base: 0 or 8

    // prologue: stage K-chunk 0 into buffer 0
    copy16_g2l(gA0, &sA[0][arow0][aseg]);
    copy16_g2l(gA1, &sA[0][arow1][aseg]);
    copy16_g2l(gB,  &sB[0][brow][aseg]);
    wait_async_lds();
    __syncthreads();

    int p = 0;
    for (int kk = 0; kk < IN_FEAT; kk += BK) {
        if (kk + BK < IN_FEAT) {
            int kn = kk + BK;
            copy16_g2l(gA0 + kn, &sA[p ^ 1][arow0][aseg]);
            copy16_g2l(gA1 + kn, &sA[p ^ 1][arow1][aseg]);
            copy16_g2l(gB  + kn, &sB[p ^ 1][brow][aseg]);
        }

        // fragments: two b128 LDS loads each, K order {kb..kb+7, kb+16..kb+23}
        v8bf a0l = *(const v8bf*)&sA[p][wm + lr][kb];
        v8bf a0h = *(const v8bf*)&sA[p][wm + lr][kb + 16];
        v8bf a1l = *(const v8bf*)&sA[p][wm + 16 + lr][kb];
        v8bf a1h = *(const v8bf*)&sA[p][wm + 16 + lr][kb + 16];
        v8bf b0l = *(const v8bf*)&sB[p][wn + lr][kb];
        v8bf b0h = *(const v8bf*)&sB[p][wn + lr][kb + 16];
        v8bf b1l = *(const v8bf*)&sB[p][wn + 16 + lr][kb];
        v8bf b1h = *(const v8bf*)&sB[p][wn + 16 + lr][kb + 16];
        v16bf a0 = CAT16(a0l, a0h);
        v16bf a1 = CAT16(a1l, a1h);
        v16bf b0 = CAT16(b0l, b0h);
        v16bf b1 = CAT16(b1l, b1h);

        c00 = __builtin_amdgcn_wmma_f32_16x16x32_bf16(false, a0, false, b0,
                                                      (short)0, c00, false, false);
        c01 = __builtin_amdgcn_wmma_f32_16x16x32_bf16(false, a0, false, b1,
                                                      (short)0, c01, false, false);
        c10 = __builtin_amdgcn_wmma_f32_16x16x32_bf16(false, a1, false, b0,
                                                      (short)0, c10, false, false);
        c11 = __builtin_amdgcn_wmma_f32_16x16x32_bf16(false, a1, false, b1,
                                                      (short)0, c11, false, false);

        wait_async_lds();
        __syncthreads();
        p ^= 1;
    }

    // D layout: lane L, element v -> row M = v + 8*(L>=16), col N = L%16
    float* zb = Z + (size_t)m0 * OUT_FEAT + n0;
    #pragma unroll
    for (int v = 0; v < 8; ++v) {
        int rowM = v + (hi << 3);
        zb[(size_t)(wm + rowM)      * OUT_FEAT + wn + lr]      = c00[v];
        zb[(size_t)(wm + rowM)      * OUT_FEAT + wn + 16 + lr] = c01[v];
        zb[(size_t)(wm + 16 + rowM) * OUT_FEAT + wn + lr]      = c10[v];
        zb[(size_t)(wm + 16 + rowM) * OUT_FEAT + wn + 16 + lr] = c11[v];
    }
}

// ---------------------------------------------------------------------------
// el/er = z @ attn_l / attn_r : one wave32 per row.
// ---------------------------------------------------------------------------
__global__ __launch_bounds__(256) void rowdot_kernel(
    const float* __restrict__ Z, const float* __restrict__ al,
    const float* __restrict__ ar, float* __restrict__ el, float* __restrict__ er)
{
    int row  = blockIdx.x * 8 + (threadIdx.x >> 5);
    int lane = threadIdx.x & 31;
    const float* zr = Z + (size_t)row * OUT_FEAT;
    float sl = 0.f, sr = 0.f;
    #pragma unroll
    for (int j = 0; j < 8; ++j) {
        int c = lane + (j << 5);
        float z = zr[c];
        sl += z * al[c];
        sr += z * ar[c];
    }
    #pragma unroll
    for (int off = 16; off > 0; off >>= 1) {
        sl += __shfl_down(sl, off, 32);
        sr += __shfl_down(sr, off, 32);
    }
    if (lane == 0) { el[row] = sl; er[row] = sr; }
}

// ---------------------------------------------------------------------------
// Edge softmax passes
// ---------------------------------------------------------------------------
__device__ __forceinline__ void atomicMaxF(float* a, float v) {
    if (v >= 0.f) atomicMax((int*)a, __float_as_int(v));
    else          atomicMin((unsigned int*)a, (unsigned int)__float_as_int(v));
}

__global__ __launch_bounds__(256) void init_md_kernel(float* m, float* den, int n) {
    int i = blockIdx.x * blockDim.x + threadIdx.x;
    if (i < n) { m[i] = -__builtin_inff(); den[i] = 0.f; }
}

__global__ __launch_bounds__(256) void edge_max_kernel(
    const int* __restrict__ src, const int* __restrict__ dst,
    const float* __restrict__ el, const float* __restrict__ er,
    float* __restrict__ ew, float* __restrict__ m, int ne)
{
    int i = blockIdx.x * blockDim.x + threadIdx.x;
    if (i >= ne) return;
    float e = el[src[i]] + er[dst[i]];
    e = (e > 0.f) ? e : NEG_SLOPE * e;
    ew[i] = e;
    atomicMaxF(m + dst[i], e);
}

__global__ __launch_bounds__(256) void edge_expsum_kernel(
    const int* __restrict__ dst, const float* __restrict__ m,
    float* __restrict__ ew, float* __restrict__ den, int ne)
{
    int i = blockIdx.x * blockDim.x + threadIdx.x;
    if (i >= ne) return;
    int d = dst[i];
    float w = __expf(ew[i] - m[d]);
    ew[i] = w;
    atomicAdd(den + d, w);
}

// One wave32 per edge: float4 gather of z[src], atomic-add scatter to out[dst].
__global__ __launch_bounds__(256) void edge_agg_kernel(
    const int* __restrict__ src, const int* __restrict__ dst,
    const float* __restrict__ ew, const float* __restrict__ den,
    const float* __restrict__ Z, float* __restrict__ out, int ne)
{
    int e = blockIdx.x * 8 + (threadIdx.x >> 5);
    if (e >= ne) return;
    int lane = threadIdx.x & 31;
    int s = src[e], d = dst[e];
    float alpha = ew[e] / (den[d] + EPS);
    const float4* zs = (const float4*)(Z + (size_t)s * OUT_FEAT);
    float* od = out + (size_t)d * OUT_FEAT;
    #pragma unroll
    for (int j = 0; j < 2; ++j) {
        int q = lane + (j << 5);               // float4 index 0..63
        float4 z4 = zs[q];
        atomicAdd(od + q * 4 + 0, alpha * z4.x);
        atomicAdd(od + q * 4 + 1, alpha * z4.y);
        atomicAdd(od + q * 4 + 2, alpha * z4.z);
        atomicAdd(od + q * 4 + 3, alpha * z4.w);
    }
}

// out[i,f] = sum_r b[r,f]
__global__ __launch_bounds__(256) void init_out_kernel(
    const float* __restrict__ b, float* __restrict__ out)
{
    int i = blockIdx.x * blockDim.x + threadIdx.x;
    int f = i & (OUT_FEAT - 1);
    out[i] = b[f] + b[OUT_FEAT + f] + b[2 * OUT_FEAT + f];
}

// ---------------------------------------------------------------------------
extern "C" void kernel_launch(void* const* d_in, const int* in_sizes, int n_in,
                              void* d_out, int out_size, void* d_ws, size_t ws_size,
                              hipStream_t stream) {
    (void)in_sizes; (void)n_in; (void)out_size; (void)ws_size;
    const float* h   = (const float*)d_in[0];
    const float* Wm  = (const float*)d_in[1];
    const float* al  = (const float*)d_in[2];
    const float* ar  = (const float*)d_in[3];
    const float* b   = (const float*)d_in[4];
    const int*   src = (const int*)d_in[5];
    const int*   dst = (const int*)d_in[6];
    float*       out = (float*)d_out;

    char* w = (char*)d_ws;
    float* Z    = (float*)w; w += (size_t)N_NODES * OUT_FEAT * sizeof(float);
    u16*   Hbf  = (u16*)w;   w += (size_t)N_NODES * IN_FEAT * sizeof(u16);
    u16*   WbfT = (u16*)w;   w += (size_t)N_REL * IN_FEAT * OUT_FEAT * sizeof(u16);
    float* el   = (float*)w; w += (size_t)N_NODES * sizeof(float);
    float* er   = (float*)w; w += (size_t)N_NODES * sizeof(float);
    float* mx   = (float*)w; w += (size_t)N_NODES * sizeof(float);
    float* den  = (float*)w; w += (size_t)N_NODES * sizeof(float);
    float* ew   = (float*)w; w += (size_t)N_EDGES * sizeof(float);

    convert_h_kernel<<<(N_NODES * IN_FEAT / 2) / 256, 256, 0, stream>>>(h, Hbf);
    convert_wT_kernel<<<(N_REL * IN_FEAT * OUT_FEAT) / 256, 256, 0, stream>>>(Wm, WbfT);
    init_out_kernel<<<(N_NODES * OUT_FEAT) / 256, 256, 0, stream>>>(b, out);

    for (int r = 0; r < N_REL; ++r) {
        init_md_kernel<<<(N_NODES + 255) / 256, 256, 0, stream>>>(mx, den, N_NODES);

        dim3 g(OUT_FEAT / BN, N_NODES / BM);           // (4, 512)
        gemm_bf16_wmma<<<g, 256, 0, stream>>>(Hbf, WbfT, Z, r);

        rowdot_kernel<<<N_NODES / 8, 256, 0, stream>>>(
            Z, al + (size_t)r * OUT_FEAT, ar + (size_t)r * OUT_FEAT, el, er);

        const int* sr = src + (size_t)r * N_EDGES;
        const int* dr = dst + (size_t)r * N_EDGES;
        edge_max_kernel<<<(N_EDGES + 255) / 256, 256, 0, stream>>>(
            sr, dr, el, er, ew, mx, N_EDGES);
        edge_expsum_kernel<<<(N_EDGES + 255) / 256, 256, 0, stream>>>(
            dr, mx, ew, den, N_EDGES);
        edge_agg_kernel<<<(N_EDGES + 7) / 8, 256, 0, stream>>>(
            sr, dr, ew, den, Z, out, N_EDGES);
    }
}
